// PatchAttention_29472065585686
// MI455X (gfx1250) — matrix operand order repaired and checked by reading, not compiled, not run on hardware
//
#include <hip/hip_runtime.h>

typedef unsigned short u16;
typedef __attribute__((ext_vector_type(16))) __bf16 v16bf;
typedef __attribute__((ext_vector_type(2)))  __bf16 v2bf;
typedef __attribute__((ext_vector_type(8)))  float  v8f;

union FragBF { v16bf v; uint4 u[2]; };

__device__ __forceinline__ u16 f2bf(float f) {
    unsigned int u = __float_as_uint(f);
    unsigned int r = 0x7FFFu + ((u >> 16) & 1u);
    return (u16)((u + r) >> 16);
}

// Pack two f32 -> packed bf16x2 (v_cvt_pk_bf16_f32 when available).
__device__ __forceinline__ unsigned pk_bf16(float lo, float hi) {
#if __has_builtin(__builtin_amdgcn_cvt_pk_bf16_f32)
    union { v2bf v; unsigned u; } cv;
    cv.v = __builtin_amdgcn_cvt_pk_bf16_f32(lo, hi);
    return cv.u;
#else
    return (unsigned)f2bf(lo) | ((unsigned)f2bf(hi) << 16);
#endif
}

__device__ __forceinline__ v8f wmma_bf16(v16bf a, v16bf b, v8f c) {
    return __builtin_amdgcn_wmma_f32_16x16x32_bf16(
        /*neg_a=*/false, a, /*neg_b=*/false, b,
        /*c_mod=*/(short)0, c, /*reuse_a=*/false, /*reuse_b=*/false);
}

// CDNA5 async copy: global -> LDS, tracked by ASYNCcnt (no VGPR round trip).
__device__ __forceinline__ void async_g2l_b128(const void* gsrc, void* ldst) {
    unsigned lds_off = (unsigned)(size_t)ldst;               // ds-relative offset
    unsigned long long gaddr = (unsigned long long)(size_t)gsrc;
    asm volatile("global_load_async_to_lds_b128 %0, %1, off"
                 :: "v"(lds_off), "v"(gaddr) : "memory");
}
__device__ __forceinline__ void wait_asynccnt0() {
    asm volatile("s_wait_asynccnt 0" ::: "memory");
}

// Problem constants
#define Bv 2
#define Nv 2048
#define Dv 1024
#define Hv 16
#define Dh 64

// ---------------------------------------------------------------------------
// Kernel A: QKV projection GEMM with WMMA bf16.
//  x[4096,1024] @ Wqkv[1024,3072] + bqkv -> Qh/Kh [BH,N,Dh] bf16, Vt [BH,Dh,N] bf16
//  Q is pre-scaled by 1/sqrt(Dh).
// ---------------------------------------------------------------------------
__global__ __launch_bounds__(128) void qkv_kernel(
    const float* __restrict__ x, const float* __restrict__ Wqkv,
    const float* __restrict__ bqkv,
    u16* __restrict__ Qh, u16* __restrict__ Kh, u16* __restrict__ Vt) {

    const int tid  = threadIdx.x;
    const int w    = tid >> 5;
    const int lane = tid & 31;
    const int t    = lane & 15;
    const int g    = lane >> 4;
    const int m0   = blockIdx.y * 64;   // row block in [0,4096)
    const int c0   = blockIdx.x * 64;   // col block in [0,3072)

    __shared__ __align__(16) u16 Ash[64 * 40];  // x tile bf16, stride 40 (80B)
    __shared__ __align__(16) u16 Wsh[64 * 40];  // W tile transposed [col][k]

    v8f acc[4];
#pragma unroll
    for (int nt = 0; nt < 4; ++nt)
#pragma unroll
        for (int r = 0; r < 8; ++r) acc[nt][r] = 0.0f;

    for (int k0 = 0; k0 < Dv; k0 += 32) {
        __syncthreads();
#pragma unroll
        for (int i = 0; i < 8; ++i) {
            int lin = tid + i * 128;              // 0..1023 (pairs)
            int ar = lin >> 4, ap = lin & 15;     // x tile: 64 rows x 16 pairs
            float2 xv = *(const float2*)&x[(size_t)(m0 + ar) * Dv + k0 + ap * 2];
            ((unsigned*)Ash)[ar * 20 + ap] = pk_bf16(xv.x, xv.y);
            int kp = lin >> 6, c = lin & 63;      // W tile: 16 k-pairs x 64 cols
            float w0 = Wqkv[(size_t)(k0 + kp * 2 + 0) * (3 * Dv) + c0 + c];
            float w1 = Wqkv[(size_t)(k0 + kp * 2 + 1) * (3 * Dv) + c0 + c];
            ((unsigned*)Wsh)[c * 20 + kp] = pk_bf16(w0, w1);
        }
        __syncthreads();

        FragBF a;
        const u16* ab = &Ash[(w * 16 + t) * 40];
        a.u[0] = *(const uint4*)(ab + g * 8);
        a.u[1] = *(const uint4*)(ab + 16 + g * 8);
#pragma unroll
        for (int nt = 0; nt < 4; ++nt) {
            FragBF bfr;
            const u16* bb = &Wsh[(nt * 16 + t) * 40 + g * 8];
            bfr.u[0] = *(const uint4*)(bb);
            bfr.u[1] = *(const uint4*)(bb + 16);
            acc[nt] = wmma_bf16(a.v, bfr.v, acc[nt]);
        }
    }

    // Epilogue: bias, scale Q, scatter to head-major bf16 buffers.
#pragma unroll
    for (int nt = 0; nt < 4; ++nt) {
        int gcol   = c0 + nt * 16 + t;          // [0,3072)
        float bias = bqkv[gcol];
        int sector = gcol >> 10;                // 0=Q 1=K 2=V
        int d  = gcol & (Dv - 1);
        int hh = d >> 6, dh = d & 63;
#pragma unroll
        for (int r = 0; r < 8; ++r) {
            int grow = m0 + w * 16 + g * 8 + r;     // [0,4096)
            int bb2  = grow >> 11;                  // batch
            int nn   = grow & (Nv - 1);
            float v  = acc[nt][r] + bias;
            if (sector == 0) v *= 0.125f;           // 1/sqrt(64)
            u16 bv = f2bf(v);
            size_t bh = (size_t)bb2 * Hv + hh;
            if (sector == 0)      Qh[(bh * Nv + nn) * Dh + dh] = bv;
            else if (sector == 1) Kh[(bh * Nv + nn) * Dh + dh] = bv;
            else                  Vt[(bh * Dh + dh) * Nv + nn] = bv;
        }
    }
}

// ---------------------------------------------------------------------------
// Kernel B: coord_proj[bh, n] = coords[b,n,:] . rel_weight[h,:]
// ---------------------------------------------------------------------------
__global__ __launch_bounds__(256) void coordproj_kernel(
    const float* __restrict__ coords, const float* __restrict__ relw,
    float* __restrict__ cp) {
    int idx = blockIdx.x * blockDim.x + threadIdx.x;   // [0, B*H*N)
    int n  = idx & (Nv - 1);
    int bh = idx >> 11;
    int h  = bh & (Hv - 1);
    int b  = bh >> 4;
    const float* cr = coords + ((size_t)b * Nv + n) * 3;
    const float* rw = relw + h * 3;
    cp[idx] = cr[0] * rw[0] + cr[1] * rw[1] + cr[2] * rw[2];
}

// ---------------------------------------------------------------------------
// Kernel C: flash attention. Block = (bh, 64 query rows), 128 threads (4 waves).
// Double-buffered K/V tiles fetched with async global->LDS DMA; the next tile's
// DMA is issued before computing the current tile (latency hidden by WMMAs).
// ---------------------------------------------------------------------------
__global__ __launch_bounds__(128) void attn_kernel(
    const u16* __restrict__ Qh, const u16* __restrict__ Kh,
    const u16* __restrict__ Vt, const float* __restrict__ cp,
    const float* __restrict__ mask, float* __restrict__ out) {

    const int tid  = threadIdx.x;
    const int w    = tid >> 5;
    const int lane = tid & 31;
    const int t    = lane & 15;
    const int g    = lane >> 4;
    const int bh   = blockIdx.y;
    const int b    = bh >> 4, h = bh & 15;
    const int i0   = blockIdx.x * 64;

    __shared__ __align__(16) u16   KshD[2][64 * 72];   // K tile [key][d]
    __shared__ __align__(16) u16   VshD[2][64 * 72];   // V^T tile [d][key]
    __shared__ __align__(16) float Ssh[4][16 * 68];    // per-wave scores f32
    __shared__ __align__(16) u16   Pbf[4][16 * 72];    // per-wave P bf16
    __shared__ float AlphaSh[4][16];
    __shared__ float LrowSh[4][16];

    // issue async DMA of one 64-key K/V tile into buffer `buf`
    auto issue_tile = [&](int j0, int buf) {
#pragma unroll
        for (int i = 0; i < 4; ++i) {
            int lin = tid + i * 128;       // 0..511
            int row = lin >> 3, cu = lin & 7;
            async_g2l_b128(Kh + ((size_t)bh * Nv + j0 + row) * Dh + cu * 8,
                           &KshD[buf][row * 72 + cu * 8]);
            async_g2l_b128(Vt + ((size_t)bh * Dh + row) * Nv + j0 + cu * 8,
                           &VshD[buf][row * 72 + cu * 8]);
        }
    };

    // Q fragments resident (Q already scaled by 1/sqrt(Dh))
    FragBF qf[2];
    {
        const u16* qb = Qh + ((size_t)bh * Nv + (i0 + w * 16 + t)) * Dh;
#pragma unroll
        for (int c = 0; c < 2; ++c) {
            qf[c].u[0] = *(const uint4*)(qb + c * 32 + g * 8);
            qf[c].u[1] = *(const uint4*)(qb + c * 32 + 16 + g * 8);
        }
    }
    const float* cpb = cp + (size_t)bh * Nv;
    float cpi[8];
#pragma unroll
    for (int r = 0; r < 8; ++r) cpi[r] = cpb[i0 + w * 16 + g * 8 + r];

    float m_run = -1.0e30f, l_run = 0.0f;
    v8f o[4];
#pragma unroll
    for (int dt = 0; dt < 4; ++dt)
#pragma unroll
        for (int r = 0; r < 8; ++r) o[dt][r] = 0.0f;

    issue_tile(0, 0);   // prime the pipeline

    for (int j0 = 0; j0 < Nv; j0 += 64) {
        const int cur = (j0 >> 6) & 1;
        wait_asynccnt0();        // this wave's DMA portion landed
        __syncthreads();         // everyone's portion landed; old buffer free
        if (j0 + 64 < Nv) issue_tile(j0 + 64, cur ^ 1);

        const u16* Ksh = KshD[cur];
        const u16* Vsh = VshD[cur];

        // S = Q K^T + cp_i - cp_j + mask  (four 16x16 tiles per wave)
#pragma unroll
        for (int jt = 0; jt < 4; ++jt) {
            v8f sa;
#pragma unroll
            for (int r = 0; r < 8; ++r) sa[r] = 0.0f;
#pragma unroll
            for (int c = 0; c < 2; ++c) {
                FragBF kb;
                const u16* kp = &Ksh[(jt * 16 + t) * 72 + c * 32 + g * 8];
                kb.u[0] = *(const uint4*)(kp);
                kb.u[1] = *(const uint4*)(kp + 16);
                sa = wmma_bf16(qf[c].v, kb.v, sa);
            }
            float cpj = cpb[j0 + jt * 16 + t];
            const float* mrow = mask + (size_t)(i0 + w * 16 + g * 8) * Nv
                                     + (j0 + jt * 16 + t);
#pragma unroll
            for (int r = 0; r < 8; ++r) {
                float sv = sa[r] + cpi[r] - cpj + mrow[(size_t)r * Nv];
                Ssh[w][(g * 8 + r) * 68 + jt * 16 + t] = sv;
            }
        }
        __syncthreads();

        // online softmax: lane handles row t, cols [g*32, g*32+32)
        float vals[32];
        {
            const float4* sp = (const float4*)(&Ssh[w][t * 68 + g * 32]);
#pragma unroll
            for (int k = 0; k < 8; ++k) {
                float4 f = sp[k];
                vals[4 * k + 0] = f.x; vals[4 * k + 1] = f.y;
                vals[4 * k + 2] = f.z; vals[4 * k + 3] = f.w;
            }
        }
        float tmax = vals[0];
#pragma unroll
        for (int k = 1; k < 32; ++k) tmax = fmaxf(tmax, vals[k]);
        tmax = fmaxf(tmax, __shfl_xor(tmax, 16, 32));
        float m_new = fmaxf(m_run, tmax);
        float alpha = __expf(m_run - m_new);
        m_run = m_new;

        union { uint4 u[4]; u16 s[32]; } pu;
        float lsum = 0.0f;
#pragma unroll
        for (int k = 0; k < 32; ++k) {
            float p = __expf(vals[k] - m_new);
            lsum += p;
            pu.s[k] = f2bf(p);
        }
        {
            uint4* pd = (uint4*)(&Pbf[w][t * 72 + g * 32]);
#pragma unroll
            for (int q = 0; q < 4; ++q) pd[q] = pu.u[q];
        }
        lsum += __shfl_xor(lsum, 16, 32);
        l_run = l_run * alpha + lsum;
        if (g == 0) AlphaSh[w][t] = alpha;
        __syncthreads();

        // rescale O accumulators by per-row alpha
#pragma unroll
        for (int r = 0; r < 8; ++r) {
            float a = AlphaSh[w][g * 8 + r];
#pragma unroll
            for (int dt = 0; dt < 4; ++dt) o[dt][r] *= a;
        }

        // O += P V  (two 32-key chunks)
#pragma unroll
        for (int c = 0; c < 2; ++c) {
            FragBF pa;
            const u16* pp = &Pbf[w][t * 72 + c * 32];
            pa.u[0] = *(const uint4*)(pp + g * 8);
            pa.u[1] = *(const uint4*)(pp + 16 + g * 8);
#pragma unroll
            for (int dt = 0; dt < 4; ++dt) {
                FragBF vb;
                const u16* vp = &Vsh[(dt * 16 + t) * 72 + c * 32 + g * 8];
                vb.u[0] = *(const uint4*)(vp);
                vb.u[1] = *(const uint4*)(vp + 16);
                o[dt] = wmma_bf16(pa.v, vb.v, o[dt]);
            }
        }
    }

    if (g == 0) LrowSh[w][t] = l_run;
    __syncthreads();

    // normalize and write out[b, i, h, dh]
#pragma unroll
    for (int r = 0; r < 8; ++r) {
        float inv = 1.0f / LrowSh[w][g * 8 + r];
        int i = i0 + w * 16 + g * 8 + r;
        float* ob = out + (((size_t)b * Nv + i) * Hv + h) * Dh;
#pragma unroll
        for (int dt = 0; dt < 4; ++dt) ob[dt * 16 + t] = o[dt][r] * inv;
    }
}

extern "C" void kernel_launch(void* const* d_in, const int* in_sizes, int n_in,
                              void* d_out, int out_size, void* d_ws, size_t ws_size,
                              hipStream_t stream) {
    (void)in_sizes; (void)n_in; (void)out_size; (void)ws_size;
    const float* x      = (const float*)d_in[0];
    const float* coords = (const float*)d_in[1];
    const float* mask   = (const float*)d_in[2];
    const float* Wqkv   = (const float*)d_in[3];
    const float* bqkv   = (const float*)d_in[4];
    const float* relw   = (const float*)d_in[5];
    float* out = (float*)d_out;

    const size_t elems = (size_t)Bv * Hv * Nv * Dh;  // 4,194,304
    u16* Qh = (u16*)d_ws;
    u16* Kh = Qh + elems;
    u16* Vt = Kh + elems;
    float* cp = (float*)(Vt + elems);

    qkv_kernel<<<dim3(48, 64), 128, 0, stream>>>(x, Wqkv, bqkv, Qh, Kh, Vt);
    coordproj_kernel<<<dim3(256), 256, 0, stream>>>(coords, relw, cp);
    attn_kernel<<<dim3(32, 32), 128, 0, stream>>>(Qh, Kh, Vt, cp, mask, out);
}
